// Decoder_7129645711887
// MI455X (gfx1250) — matrix operand order repaired and checked by reading, not compiled, not run on hardware
//
#include <hip/hip_runtime.h>

#define B_TOT 32768
#define L_POS 32
#define D_DIM 512
#define H_DIM 256
#define O_DIM 1000
#define O_PAD 1008
#define BT    32            // batch rows per block
#define LN_EPS 1e-5f

// strides (elements) chosen so every 16-byte fragment chunk is 16B aligned
#define SA_STRIDE 520       // ushort, 1040 B/row (=65*16)
#define HF_STRIDE 260       // float,  1040 B/row (same storage as sA)
#define SH_STRIDE 264       // ushort,  528 B/row (=33*16)

typedef __attribute__((ext_vector_type(16))) __bf16 v16bf;
typedef __attribute__((ext_vector_type(8)))  float  v8f;

union Frag16 { v16bf v; uint4 u[2]; };

__device__ __forceinline__ unsigned short f2bf(float f) {
  union { float f; unsigned int u; } c; c.f = f;
  unsigned int r = c.u + 0x7FFFu + ((c.u >> 16) & 1u);   // round-to-nearest-even
  return (unsigned short)(r >> 16);
}

// ---------------------------------------------------------------------------
// Prep: W1 (D x H row-major fp32) -> W1T bf16 column-major [H][D]
//       W2 (H x O row-major fp32) -> W2T bf16 column-major [O_PAD][H], zero pad
// ---------------------------------------------------------------------------
__global__ void prep_weights(const float* __restrict__ W1,
                             const float* __restrict__ W2,
                             unsigned short* __restrict__ W1T,
                             unsigned short* __restrict__ W2T) {
  int i = blockIdx.x * blockDim.x + threadIdx.x;
  if (i < D_DIM * H_DIM) {
    int h = i / D_DIM, d = i - h * D_DIM;           // W1T[h][d]
    W1T[i] = f2bf(W1[d * H_DIM + h]);
  }
  int j = i - D_DIM * H_DIM;
  if (j >= 0 && j < O_PAD * H_DIM) {
    int o = j / H_DIM, h = j - o * H_DIM;           // W2T[o][h]
    W2T[j] = (o < O_DIM) ? f2bf(W2[h * O_DIM + o]) : (unsigned short)0;
  }
}

// ---------------------------------------------------------------------------
// Fused: gather -> GEMM1(bf16 WMMA) -> bias/ReLU -> LayerNorm -> GEMM2 -> +b2
// 256 threads = 8 waves; wave = (mt in 0..1) x (nq in 0..3); 32 rows/block
// ---------------------------------------------------------------------------
__global__ __launch_bounds__(256) void fused_decoder(
    const int* __restrict__ z_h,
    const float* __restrict__ tables,
    const float* __restrict__ b1,
    const float* __restrict__ gamma,
    const float* __restrict__ beta,
    const float* __restrict__ b2,
    const unsigned short* __restrict__ W1T,
    const unsigned short* __restrict__ W2T,
    float* __restrict__ logits,
    float* __restrict__ xh)
{
  __shared__ __align__(16) unsigned short sA[BT * SA_STRIDE];  // 33280 B (reused as fp32 h)
  __shared__ __align__(16) unsigned short sH[BT * SH_STRIDE];  // 16896 B (bf16 hn)
  __shared__ unsigned char sZ[BT * L_POS];                     //  1024 B

  const int t     = threadIdx.x;
  const int gbase = blockIdx.x * BT;

  // ---- phase 0: stage ternary indices (z+1 in {0,1,2}) --------------------
  for (int i = t; i < BT * L_POS; i += 256) {
    int row = i >> 5, l = i & 31;
    sZ[i] = (unsigned char)(z_h[(gbase + row) * L_POS + l] + 1);
  }
  __syncthreads();

  // ---- phase 1: gather-mask-accumulate -> sA(bf16) + xh(fp32) -------------
  // chunk = (row, 4 cols); lanes of a wave share a row -> scalar mask branch
  for (int c = t; c < BT * (D_DIM / 4); c += 256) {
    int row = c >> 7;
    int col = c & 127;
    float4 acc = {0.f, 0.f, 0.f, 0.f};
    const unsigned char* zr = &sZ[row * L_POS];
    for (int l = 0; l < L_POS; ++l) {
      unsigned int u = zr[l];
      if (u) {
        const float4 e = *(const float4*)(tables + (size_t)((l * 3 + u) * D_DIM) + col * 4);
        acc.x += e.x; acc.y += e.y; acc.z += e.z; acc.w += e.w;
      }
    }
    *(float4*)(xh + (size_t)(gbase + row) * D_DIM + col * 4) = acc;
    unsigned short* dst = &sA[row * SA_STRIDE + col * 4];
    dst[0] = f2bf(acc.x); dst[1] = f2bf(acc.y);
    dst[2] = f2bf(acc.z); dst[3] = f2bf(acc.w);
  }
  __syncthreads();

  // ---- wave coordinates ---------------------------------------------------
  const int lane = t & 31;
  const int wave = t >> 5;
  const int mt   = wave & 1;        // M tile (16 rows)
  const int nq   = wave >> 1;       // N quarter
  const int m0   = mt * 16;
  const int ln   = lane & 15;
  const int hi   = lane >> 4;
  const int arow = m0 + ln;         // A-fragment source row for this lane

  // ---- phase 2: GEMM1 x_h @ W1 (K=512, 16 wmma steps, 4 N-tiles/wave) -----
  v8f acc1[4];
#pragma unroll
  for (int i = 0; i < 4; ++i) acc1[i] = (v8f){0.f,0.f,0.f,0.f,0.f,0.f,0.f,0.f};

  for (int ks = 0; ks < D_DIM / 32; ++ks) {
    Frag16 a;
    const int ka = ks * 32 + 8 * hi;                 // A: K 0-7 / 8-15 per half-wave
    a.u[0] = *(const uint4*)&sA[arow * SA_STRIDE + ka];
    a.u[1] = *(const uint4*)&sA[arow * SA_STRIDE + ka + 16];
    const int kb = ks * 32 + 8 * hi;                 // B: K 0-7 / 8-15 per half-wave
#pragma unroll
    for (int nt4 = 0; nt4 < 4; ++nt4) {
      const int n = (nq * 4 + nt4) * 16 + ln;
      Frag16 b;
      const unsigned short* bp = W1T + (size_t)n * D_DIM + kb;
      b.u[0] = *(const uint4*)(bp);
      b.u[1] = *(const uint4*)(bp + 16);
      acc1[nt4] = __builtin_amdgcn_wmma_f32_16x16x32_bf16(
          false, a.v, false, b.v, (short)0, acc1[nt4], false, false);
    }
  }
  __syncthreads();                                   // all sA reads retired

  // ---- phase 2b: bias + ReLU -> hF (reuses sA storage as fp32) ------------
  float* hF = reinterpret_cast<float*>(sA);
#pragma unroll
  for (int nt4 = 0; nt4 < 4; ++nt4) {
    const int col  = (nq * 4 + nt4) * 16 + ln;       // C/D: N = lane&15
    const float bias = b1[col];
#pragma unroll
    for (int r = 0; r < 8; ++r) {                    // C/D: M = 8*hi + r
      float v = acc1[nt4][r] + bias;
      hF[(m0 + 8 * hi + r) * HF_STRIDE + col] = v > 0.f ? v : 0.f;
    }
  }
  __syncthreads();

  // ---- phase 3: LayerNorm per row -> sH (bf16), wave32 shuffle reduce -----
  {
    float g[8], be[8];
#pragma unroll
    for (int j = 0; j < 8; ++j) {
      g[j]  = gamma[lane + 32 * j];
      be[j] = beta [lane + 32 * j];
    }
    for (int rr = 0; rr < 4; ++rr) {                 // 8 waves x 4 rows = 32
      const int row = wave * 4 + rr;
      float v[8], s = 0.f, sq = 0.f;
#pragma unroll
      for (int j = 0; j < 8; ++j) {
        v[j] = hF[row * HF_STRIDE + lane + 32 * j];
        s += v[j]; sq += v[j] * v[j];
      }
#pragma unroll
      for (int off = 16; off > 0; off >>= 1) {
        s  += __shfl_xor(s,  off, 32);
        sq += __shfl_xor(sq, off, 32);
      }
      const float mu   = s * (1.f / H_DIM);
      const float var  = sq * (1.f / H_DIM) - mu * mu;
      const float rstd = rsqrtf(var + LN_EPS);
#pragma unroll
      for (int j = 0; j < 8; ++j) {
        sH[row * SH_STRIDE + lane + 32 * j] = f2bf((v[j] - mu) * rstd * g[j] + be[j]);
      }
    }
  }
  __syncthreads();

  // ---- phase 4: GEMM2 hn @ W2 + b2 (K=256, 8 wmma steps, 63 N-tiles) ------
  const int ntStart = nq * 16;
  const int ntEnd   = (nq == 3) ? 63 : (ntStart + 16);
  for (int nt = ntStart; nt < ntEnd; ++nt) {
    v8f acc = (v8f){0.f,0.f,0.f,0.f,0.f,0.f,0.f,0.f};
    for (int ks = 0; ks < H_DIM / 32; ++ks) {
      Frag16 a;
      const int ka = ks * 32 + 8 * hi;
      a.u[0] = *(const uint4*)&sH[arow * SH_STRIDE + ka];
      a.u[1] = *(const uint4*)&sH[arow * SH_STRIDE + ka + 16];
      Frag16 b;
      const int n = nt * 16 + ln;
      const unsigned short* bp = W2T + (size_t)n * H_DIM + ks * 32 + 8 * hi;
      b.u[0] = *(const uint4*)(bp);
      b.u[1] = *(const uint4*)(bp + 16);
      acc = __builtin_amdgcn_wmma_f32_16x16x32_bf16(
          false, a.v, false, b.v, (short)0, acc, false, false);
    }
    const int o = nt * 16 + ln;
    if (o < O_DIM) {
      const float bias = b2[o];
#pragma unroll
      for (int r = 0; r < 8; ++r) {
        logits[(size_t)(gbase + m0 + 8 * hi + r) * O_DIM + o] = acc[r] + bias;
      }
    }
  }
}

// ---------------------------------------------------------------------------
extern "C" void kernel_launch(void* const* d_in, const int* in_sizes, int n_in,
                              void* d_out, int out_size, void* d_ws, size_t ws_size,
                              hipStream_t stream) {
  (void)in_sizes; (void)n_in; (void)out_size; (void)ws_size;
  const int*   z      = (const int*)  d_in[0];
  const float* tables = (const float*)d_in[1];
  const float* W1     = (const float*)d_in[2];
  const float* b1     = (const float*)d_in[3];
  const float* gamma  = (const float*)d_in[4];
  const float* beta   = (const float*)d_in[5];
  const float* W2     = (const float*)d_in[6];
  const float* b2     = (const float*)d_in[7];

  float* logits = (float*)d_out;                          // [B, O]
  float* xh     = logits + (size_t)B_TOT * O_DIM;         // [B, D]

  unsigned short* W1T = (unsigned short*)d_ws;            // 256 KB
  unsigned short* W2T = W1T + D_DIM * H_DIM;              // 516 KB

  const int prep_n = D_DIM * H_DIM + O_PAD * H_DIM;       // 389120
  prep_weights<<<(prep_n + 255) / 256, 256, 0, stream>>>(W1, W2, W1T, W2T);

  fused_decoder<<<B_TOT / BT, 256, 0, stream>>>(
      z, tables, b1, gamma, beta, b2, W1T, W2T, logits, xh);
}
